// MultiHeadAttention_10110353014942
// MI455X (gfx1250) — matrix operand order repaired and checked by reading, not compiled, not run on hardware
//
#include <hip/hip_runtime.h>

#define DEV __device__ __forceinline__

typedef __bf16 bf16_t;
typedef __attribute__((ext_vector_type(16))) __bf16 v16bf;
typedef __attribute__((ext_vector_type(8)))  __bf16 v8bf;
typedef __attribute__((ext_vector_type(8)))  float   v8f;

static constexpr int BB = 4, SS = 2048, DD = 1024, HH = 16;
static constexpr float SCALE_F = 0.125f;          // 1/sqrt(64)
static constexpr int M_TOT = BB * SS;             // 8192
static constexpr int N_TOT = 1024;
static constexpr int K_TOT = 1024;

// ---------------- WMMA helpers (CDNA5 16x16x32 bf16, f32 accum) ----------------

DEV v8f wmma_bf16(v16bf a, v16bf b, v8f c) {
  // (neg_a, A, neg_b, B, c_mod, C, reuse_a, reuse_b)
  return __builtin_amdgcn_wmma_f32_16x16x32_bf16(false, a, false, b, (short)0, c,
                                                 false, false);
}

// A-matrix fragment, 16x32 bf16, row-major source with element stride `stride`.
// ISA layout: lanes 0-15 row=lane, K={kh..kh+7, 16+kh..16+kh+7}, kh = (lane>>4)*8.
DEV v16bf frag_a(const bf16_t* base, int stride) {
  const int lane = threadIdx.x & 31;
  const int row  = lane & 15;
  const int kh   = (lane >> 4) << 3;
  const bf16_t* p = base + row * stride;
  v8bf lo = *(const v8bf*)(p + kh);
  v8bf hi = *(const v8bf*)(p + 16 + kh);
  v16bf r;
#pragma unroll
  for (int i = 0; i < 8; ++i) { r[i] = lo[i]; r[i + 8] = hi[i]; }
  return r;
}

// B-matrix fragment 32x16, sourced from B^T (n-major, k-contiguous, stride elems).
// ISA layout: lane col = lane&15, K={kh..kh+7, 16+kh..16+kh+7}, kh = (lane>>4)*8.
DEV v16bf frag_b(const bf16_t* base, int stride) {
  const int lane = threadIdx.x & 31;
  const int col  = lane & 15;
  const int kh   = (lane >> 4) << 3;
  const bf16_t* p = base + col * stride;
  v8bf lo = *(const v8bf*)(p + kh);
  v8bf hi = *(const v8bf*)(p + 16 + kh);
  v16bf r;
#pragma unroll
  for (int i = 0; i < 8; ++i) { r[i] = lo[i]; r[i + 8] = hi[i]; }
  return r;
}

DEV unsigned short f2bf_bits(float f) {
  unsigned u = __float_as_uint(f);
  u += 0x7FFFu + ((u >> 16) & 1u);   // round-to-nearest-even
  return (unsigned short)(u >> 16);
}
DEV bf16_t f2bf(float f) {
  union { unsigned short u; bf16_t b; } c;
  c.u = f2bf_bits(f);
  return c.b;
}

// ---------------- gfx1250 async global->LDS copy (ASYNCcnt path) ---------------
// Per CDNA5 ISA: GLOBAL_LOAD_ASYNC_TO_LDS_B128 copies 16B per
// active lane directly into LDS, tracked by ASYNCcnt. VDST VGPR holds the
// wave-relative LDS byte offset; a generic shared pointer's low 32 bits are
// exactly that offset (flat LDS aperture keeps the offset in addr[31:0]).

DEV void async_copy_b128(const bf16_t* gsrc, bf16_t* ldst) {
  unsigned lds_off = (unsigned)(unsigned long long)ldst;
  asm volatile("global_load_async_to_lds_b128 %0, %1, off"
               :
               : "v"(lds_off), "v"(gsrc)
               : "memory");
}

DEV void wait_asynccnt0() {
  asm volatile("s_wait_asynccnt 0" ::: "memory");
}

// ---------------- stage 0: dtype staging ----------------

__global__ void mha_cvt_bf16_kernel(const float* __restrict__ src,
                                    unsigned short* __restrict__ dst, int n) {
  for (int i = blockIdx.x * blockDim.x + threadIdx.x; i < n;
       i += gridDim.x * blockDim.x)
    dst[i] = f2bf_bits(src[i]);
}

__global__ void mha_copy_f32_kernel(const float* __restrict__ src,
                                    float* __restrict__ dst, int n) {
  for (int i = blockIdx.x * blockDim.x + threadIdx.x; i < n;
       i += gridDim.x * blockDim.x)
    dst[i] = src[i];
}

// ---------------- stage 1/3: 8192x1024x1024 GEMM, C = A @ W^T + bias ----------
// A: bf16 [M,K] row-major.  W: bf16 [N,K] row-major (== B^T).
// Double-buffered LDS tiles filled by async global->LDS DMA; WMMA consumes the
// live buffer while the next K-tile streams into the other one.
// MODE 0: out bf16 [B,H,S,64]      (Q/K projections)
// MODE 1: out bf16 [B,H,64,S]      (V projection, pre-transposed for P@V)
// MODE 2: out f32  [M,N] row-major (output projection)

template <int MODE>
__launch_bounds__(256)
__global__ void mha_gemm_kernel(const bf16_t* __restrict__ A,
                                const bf16_t* __restrict__ W,
                                const float* __restrict__ bias,
                                void* __restrict__ out) {
  __shared__ alignas(16) bf16_t At[2][128][56];  // 112B rows: conflict-tidy
  __shared__ alignas(16) bf16_t Wt[2][128][56];
  const int tid  = threadIdx.x;
  const int lane = tid & 31;
  const int wave = tid >> 5;
  const int wm   = wave >> 1;           // 0..3  (32-row slice)
  const int wn   = wave & 1;            // 0..1  (64-col slice)
  const int gm0  = blockIdx.x * 128;
  const int gn0  = blockIdx.y * 128;

  v8f acc[2][4];
#pragma unroll
  for (int i = 0; i < 2; ++i)
#pragma unroll
    for (int j = 0; j < 4; ++j) acc[i][j] = (v8f)(0.0f);

  const int lrow  = tid >> 1;           // 0..127
  const int lhalf = (tid & 1) * 16;     // 0 or 16

  // issue one K-tile (128x32 of A and W) as 4 async 16B copies per thread
  auto issue_tile = [&](int buf, int kt) {
    const bf16_t* ap = A + (size_t)(gm0 + lrow) * K_TOT + kt + lhalf;
    const bf16_t* wp = W + (size_t)(gn0 + lrow) * K_TOT + kt + lhalf;
    async_copy_b128(ap,     &At[buf][lrow][lhalf]);
    async_copy_b128(ap + 8, &At[buf][lrow][lhalf + 8]);
    async_copy_b128(wp,     &Wt[buf][lrow][lhalf]);
    async_copy_b128(wp + 8, &Wt[buf][lrow][lhalf + 8]);
  };

  issue_tile(0, 0);
  wait_asynccnt0();
  __syncthreads();

  int cur = 0;
  for (int kt = 0; kt < K_TOT; kt += 32) {
    const bool has_next = (kt + 32 < K_TOT);
    if (has_next) issue_tile(cur ^ 1, kt + 32);   // overlap DMA with WMMA

    v16bf af0 = frag_a(&At[cur][wm * 32][0], 56);
    v16bf af1 = frag_a(&At[cur][wm * 32 + 16][0], 56);
#pragma unroll
    for (int j = 0; j < 4; ++j) {
      v16bf bfj = frag_b(&Wt[cur][wn * 64 + 16 * j][0], 56);
      acc[0][j] = wmma_bf16(af0, bfj, acc[0][j]);
      acc[1][j] = wmma_bf16(af1, bfj, acc[1][j]);
    }

    if (has_next) wait_asynccnt0();   // own async copies landed
    __syncthreads();                  // everyone's copies landed / reads done
    cur ^= 1;
  }

  // C layout: lane col = lane&15, VGPR r -> row (lane>>4)*8 + r
  const int mb = (lane >> 4) << 3;
  const int nc = lane & 15;
#pragma unroll
  for (int i = 0; i < 2; ++i) {
#pragma unroll
    for (int j = 0; j < 4; ++j) {
      const int gn = gn0 + wn * 64 + 16 * j + nc;
      const float bv = bias[gn];
#pragma unroll
      for (int r = 0; r < 8; ++r) {
        const int gm = gm0 + wm * 32 + 16 * i + mb + r;
        const float v = acc[i][j][r] + bv;
        if (MODE == 2) {
          ((float*)out)[(size_t)gm * N_TOT + gn] = v;
        } else {
          const int b = gm >> 11, s = gm & 2047;
          const int h = gn >> 6,  dk = gn & 63;
          bf16_t* o = (bf16_t*)out;
          if (MODE == 0)
            o[(((size_t)b * HH + h) * SS + s) * 64 + dk] = f2bf(v);
          else
            o[(((size_t)b * HH + h) * 64 + dk) * SS + s] = f2bf(v);
        }
      }
    }
  }
}

// ---------------- stage 2: flash attention ----------------
// grid (S/64, B*H), 128 threads = 4 independent waves; wave owns 16 queries.

__launch_bounds__(128)
__global__ void mha_attention_kernel(const bf16_t* __restrict__ xq,
                                     const bf16_t* __restrict__ xk,
                                     const bf16_t* __restrict__ xvT,
                                     const unsigned char* __restrict__ mask,
                                     bf16_t* __restrict__ ctx) {
  __shared__ alignas(16) bf16_t pbuf[4][16][56];   // per-wave P scratch
  const int lane = threadIdx.x & 31;
  const int wave = threadIdx.x >> 5;
  const int bh = blockIdx.y;
  const int b  = bh >> 4;
  const int h  = bh & 15;
  const int q0 = blockIdx.x * 64 + wave * 16;

  const bf16_t* Qp = xq  + ((size_t)bh * SS + q0) * 64;
  const bf16_t* Kp = xk  + (size_t)bh * SS * 64;
  const bf16_t* Vt = xvT + (size_t)bh * 64 * SS;
  const unsigned char* mp = mask + ((size_t)b * SS + q0) * SS;

  // Q fragments held in registers for the whole key loop (16x64 = 2 frags)
  v16bf qf0 = frag_a(Qp, 64);
  v16bf qf1 = frag_a(Qp + 32, 64);

  const int mb = (lane >> 4) << 3;
  const int nc = lane & 15;

  float mrow[8], lrow[8];
  v8f acc[4];
#pragma unroll
  for (int r = 0; r < 8; ++r) { mrow[r] = -3.0e38f; lrow[r] = 0.0f; }
#pragma unroll
  for (int j = 0; j < 4; ++j) acc[j] = (v8f)(0.0f);

  bf16_t* myp = &pbuf[wave][0][0];

  for (int kb = 0; kb < SS; kb += 32) {
    // scores S = Q(16x64) @ K^T(64x32); B^T == K row-major
    v8f s0 = (v8f)(0.0f), s1 = (v8f)(0.0f);
    {
      v16bf k0a = frag_b(Kp + (size_t)kb * 64, 64);
      v16bf k0b = frag_b(Kp + (size_t)kb * 64 + 32, 64);
      s0 = wmma_bf16(qf0, k0a, s0);
      s0 = wmma_bf16(qf1, k0b, s0);
      v16bf k1a = frag_b(Kp + (size_t)(kb + 16) * 64, 64);
      v16bf k1b = frag_b(Kp + (size_t)(kb + 16) * 64 + 32, 64);
      s1 = wmma_bf16(qf0, k1a, s1);
      s1 = wmma_bf16(qf1, k1b, s1);
    }

    // scale + pad mask (coalesced byte loads: lanes sweep contiguous k)
    float x0[8], x1[8];
#pragma unroll
    for (int r = 0; r < 8; ++r) {
      const unsigned char* mr = mp + (size_t)(mb + r) * SS + kb + nc;
      float a = s0[r] * SCALE_F;
      float c = s1[r] * SCALE_F;
      if (!mr[0])  a = -1.0e9f;
      if (!mr[16]) c = -1.0e9f;
      x0[r] = a; x1[r] = c;
    }

    // online softmax: reductions over 16 cols live in each 16-lane half
#pragma unroll
    for (int r = 0; r < 8; ++r) {
      float cm = fmaxf(x0[r], x1[r]);
      cm = fmaxf(cm, __shfl_xor(cm, 8, 32));
      cm = fmaxf(cm, __shfl_xor(cm, 4, 32));
      cm = fmaxf(cm, __shfl_xor(cm, 2, 32));
      cm = fmaxf(cm, __shfl_xor(cm, 1, 32));
      const float mnew  = fmaxf(mrow[r], cm);
      const float alpha = __expf(mrow[r] - mnew);
      const float p0 = __expf(x0[r] - mnew);
      const float p1 = __expf(x1[r] - mnew);
      float srow = p0 + p1;
      srow += __shfl_xor(srow, 8, 32);
      srow += __shfl_xor(srow, 4, 32);
      srow += __shfl_xor(srow, 2, 32);
      srow += __shfl_xor(srow, 1, 32);
      mrow[r] = mnew;
      lrow[r] = lrow[r] * alpha + srow;
#pragma unroll
      for (int j = 0; j < 4; ++j) acc[j][r] *= alpha;
      // stash P in A-fragment-friendly row-major LDS (wave-private, in-order DS)
      myp[(mb + r) * 56 + nc]      = f2bf(p0);
      myp[(mb + r) * 56 + nc + 16] = f2bf(p1);
    }

    // ctx += P(16x32) @ V(32x64); B^T == V^T rows (dv-major, s-contiguous)
    v16bf pf = frag_a(myp, 56);
#pragma unroll
    for (int j = 0; j < 4; ++j) {
      v16bf vf = frag_b(Vt + (size_t)(j * 16) * SS + kb, SS);
      acc[j] = wmma_bf16(pf, vf, acc[j]);
    }
  }

  // normalize and emit ctx as [B*S, H*64] bf16 (A matrix of output projection)
  bf16_t* op = ctx + ((size_t)b * SS + q0) * 1024 + h * 64;
#pragma unroll
  for (int r = 0; r < 8; ++r) {
    const float inv = 1.0f / lrow[r];
#pragma unroll
    for (int j = 0; j < 4; ++j)
      op[(size_t)(mb + r) * 1024 + j * 16 + nc] = f2bf(acc[j][r] * inv);
  }
}

// ---------------- host-side orchestration ----------------

extern "C" void kernel_launch(void* const* d_in, const int* in_sizes, int n_in,
                              void* d_out, int out_size, void* d_ws, size_t ws_size,
                              hipStream_t stream) {
  (void)in_sizes; (void)n_in; (void)out_size; (void)ws_size;
  const float* q  = (const float*)d_in[0];
  const float* k  = (const float*)d_in[1];
  const float* v  = (const float*)d_in[2];
  const unsigned char* mask = (const unsigned char*)d_in[3];
  const float* wq = (const float*)d_in[4];
  const float* bq = (const float*)d_in[5];
  const float* wk = (const float*)d_in[6];
  const float* bk = (const float*)d_in[7];
  const float* wv = (const float*)d_in[8];
  const float* bv = (const float*)d_in[9];
  const float* wo = (const float*)d_in[10];
  const float* bo = (const float*)d_in[11];

  const size_t nIn = (size_t)BB * SS * DD;   // 8,388,608 elements
  const size_t nW  = (size_t)DD * DD;        // 1,048,576 elements

  size_t off = 0;
  auto grab = [&](size_t bytes) {
    void* p = (char*)d_ws + off;
    off += (bytes + 255) & ~(size_t)255;
    return p;
  };
  unsigned short* qb  = (unsigned short*)grab(nIn * 2);
  unsigned short* kb  = (unsigned short*)grab(nIn * 2);
  unsigned short* vb  = (unsigned short*)grab(nIn * 2);
  unsigned short* wqb = (unsigned short*)grab(nW * 2);
  unsigned short* wkb = (unsigned short*)grab(nW * 2);
  unsigned short* wvb = (unsigned short*)grab(nW * 2);
  unsigned short* wob = (unsigned short*)grab(nW * 2);
  bf16_t* xq  = (bf16_t*)grab(nIn * 2);
  bf16_t* xk  = (bf16_t*)grab(nIn * 2);
  bf16_t* xvT = (bf16_t*)grab(nIn * 2);
  bf16_t* ctx = (bf16_t*)qb;   // qb is dead after the Q projection

  // stage 0: bf16 staging + q passthrough output
  mha_cvt_bf16_kernel<<<2048, 256, 0, stream>>>(q, qb, (int)nIn);
  mha_cvt_bf16_kernel<<<2048, 256, 0, stream>>>(k, kb, (int)nIn);
  mha_cvt_bf16_kernel<<<2048, 256, 0, stream>>>(v, vb, (int)nIn);
  mha_cvt_bf16_kernel<<<512, 256, 0, stream>>>(wq, wqb, (int)nW);
  mha_cvt_bf16_kernel<<<512, 256, 0, stream>>>(wk, wkb, (int)nW);
  mha_cvt_bf16_kernel<<<512, 256, 0, stream>>>(wv, wvb, (int)nW);
  mha_cvt_bf16_kernel<<<512, 256, 0, stream>>>(wo, wob, (int)nW);
  mha_copy_f32_kernel<<<2048, 256, 0, stream>>>(q, (float*)d_out + nIn, (int)nIn);

  // stage 1: projections (Q,K -> [B,H,S,64]; V -> [B,H,64,S])
  dim3 gg(M_TOT / 128, N_TOT / 128);
  mha_gemm_kernel<0><<<gg, 256, 0, stream>>>((const bf16_t*)qb, (const bf16_t*)wqb, bq, xq);
  mha_gemm_kernel<0><<<gg, 256, 0, stream>>>((const bf16_t*)kb, (const bf16_t*)wkb, bk, xk);
  mha_gemm_kernel<1><<<gg, 256, 0, stream>>>((const bf16_t*)vb, (const bf16_t*)wvb, bv, xvT);

  // stage 2: flash attention -> ctx [B*S, H*64] bf16
  mha_attention_kernel<<<dim3(SS / 64, BB * HH), 128, 0, stream>>>(xq, xk, xvT, mask, ctx);

  // stage 3: output projection -> f32 d_out
  mha_gemm_kernel<2><<<gg, 256, 0, stream>>>(ctx, (const bf16_t*)wob, bo, d_out);
}
